// deformable_self_attention_42769284333657
// MI455X (gfx1250) — compile-verified, matrix operand order checked by reading, not run on hardware
//
#include <hip/hip_runtime.h>

typedef __attribute__((ext_vector_type(16))) _Float16 v16h;
typedef __attribute__((ext_vector_type(8)))  _Float16 v8h;
typedef __attribute__((ext_vector_type(8)))  float    v8f;

// Problem constants
#define BATCH   2
#define QD      128
#define NQ      (QD * QD)          // 16384
#define MTOT    (BATCH * NQ)       // 32768
#define EMBED   256
#define HEADS   8
#define POINTS  4
#define DHEAD   32
#define FFN     512
#define NPROJ   352                // 256 (val) + 64 (off) + 32 (attn)

// ---------------------------------------------------------------------------
// Elementwise helpers
// ---------------------------------------------------------------------------
__global__ void cvt_f32_to_f16(const float* __restrict__ src,
                               _Float16* __restrict__ dst, int n) {
  int i = blockIdx.x * blockDim.x + threadIdx.x;
  if (i < n) dst[i] = (_Float16)src[i];
}

// src is (Kdim x Ncols) row-major f32; dst is (Ncols x Kdim) row-major f16.
__global__ void transpose_cvt(const float* __restrict__ src,
                              _Float16* __restrict__ dst,
                              int Kdim, int Ncols) {
  int i = blockIdx.x * blockDim.x + threadIdx.x;
  if (i < Kdim * Ncols) {
    int k = i / Ncols;
    int n = i - k * Ncols;
    dst[(size_t)n * Kdim + k] = (_Float16)src[i];
  }
}

__global__ void pack_bias(const float* __restrict__ bv,
                          const float* __restrict__ bo,
                          const float* __restrict__ ba,
                          float* __restrict__ dst) {
  int t = blockIdx.x * blockDim.x + threadIdx.x;
  if (t < 256)      dst[t] = bv[t];
  else if (t < 320) dst[t] = bo[t - 256];
  else if (t < 352) dst[t] = ba[t - 320];
}

// ---------------------------------------------------------------------------
// WMMA fragment load: 16-bit A-matrix 16x32 layout (CDNA5 ISA 7.12.2).
// lane<16 : row = lane,   holds K[0..7] in v0..3 and K[16..23] in v4..7
// lane>=16: row = lane-16, holds K[8..15] in v0..3 and K[24..31] in v4..7
// `p` must point at rowbase + k0 + ((lane>>4)*8).
// ---------------------------------------------------------------------------
__device__ __forceinline__ v16h load_frag(const _Float16* p) {
  v8h lo = *(const v8h*)(p);
  v8h hi = *(const v8h*)(p + 16);
  v16h r;
#pragma unroll
  for (int i = 0; i < 8; ++i) { r[i] = lo[i]; r[8 + i] = hi[i]; }
  return r;
}

// ---------------------------------------------------------------------------
// Generic WMMA GEMM:  C(M x N) = A(M x K) * Bt(N x K)^T  (+ epilogue MODE)
//   MODE 0: proj    -> n<256: value_h f16 (stride 256); else offattn f32 (stride 96)
//   MODE 1: outproj -> v += bias + 2*q ; x_f32 + x_f16 (stride 256)
//   MODE 2: ffn1    -> relu(v + bias) -> f16 (stride N)
//   MODE 3: ffn2    -> v + bias -> f32 (stride 256)
// Block: 256 threads = 8 waves; each wave owns a 32x32 tile; block = 256M x 32N.
// ---------------------------------------------------------------------------
template <int K, int MODE>
__global__ __launch_bounds__(256) void gemm_wmma(
    const _Float16* __restrict__ A, const _Float16* __restrict__ Bt,
    const float* __restrict__ bias, int N,
    const float* __restrict__ aux,      // MODE1: original q (f32)
    _Float16* __restrict__ outh, float* __restrict__ outf) {
  const int lane  = threadIdx.x & 31;
  const int wave  = threadIdx.x >> 5;
  const int n0    = blockIdx.x * 32;
  const int m0    = blockIdx.y * 256 + wave * 32;
  const int lrow  = lane & 15;
  const int khalf = (lane >> 4) << 3;

  const _Float16* pa0 = A  + (size_t)(m0 + lrow)      * K + khalf;
  const _Float16* pa1 = A  + (size_t)(m0 + 16 + lrow) * K + khalf;
  const _Float16* pb0 = Bt + (size_t)(n0 + lrow)      * K + khalf;
  const _Float16* pb1 = Bt + (size_t)(n0 + 16 + lrow) * K + khalf;

  v8f acc[2][2] = {};

  for (int k0 = 0; k0 < K; k0 += 32) {
    if (k0 + 32 < K) {               // emits global_prefetch_b8
      __builtin_prefetch(pa0 + k0 + 32, 0, 0);
      __builtin_prefetch(pb0 + k0 + 32, 0, 0);
    }
    v16h af[2], bf[2];
    af[0] = load_frag(pa0 + k0);
    af[1] = load_frag(pa1 + k0);
    bf[0] = load_frag(pb0 + k0);
    bf[1] = load_frag(pb1 + k0);
#pragma unroll
    for (int i = 0; i < 2; ++i)
#pragma unroll
      for (int j = 0; j < 2; ++j)
        acc[i][j] = __builtin_amdgcn_wmma_f32_16x16x32_f16(
            false, af[i], false, bf[j], (short)0, acc[i][j], false, false);
  }

  // Epilogue. C/D layout: lane<16 -> N=lane, M=r ; lane>=16 -> N=lane-16, M=r+8.
  const int ncol   = lane & 15;
  const int rowoff = (lane >> 4) * 8;
#pragma unroll
  for (int j = 0; j < 2; ++j) {
    const int n  = n0 + j * 16 + ncol;
    const float bv = bias[n];
#pragma unroll
    for (int i = 0; i < 2; ++i) {
      const int mbase = m0 + i * 16 + rowoff;
#pragma unroll
      for (int r = 0; r < 8; ++r) {
        const size_t m = (size_t)(mbase + r);
        float v = acc[i][j][r] + bv;
        if (MODE == 0) {
          if (n < 256) outh[m * 256 + n] = (_Float16)v;
          else         outf[m * 96 + (n - 256)] = v;
        } else if (MODE == 1) {
          v += 2.0f * aux[m * 256 + n];
          outf[m * 256 + n] = v;
          outh[m * 256 + n] = (_Float16)v;
        } else if (MODE == 2) {
          v = v > 0.0f ? v : 0.0f;
          outh[m * (size_t)N + n] = (_Float16)v;
        } else {
          outf[m * 256 + n] = v;
        }
      }
    }
  }
}

// ---------------------------------------------------------------------------
// Deformable sampling. One wave per (m, head); lane = channel d in [0,32).
// value_h: (MTOT x 256) f16, col = h*32 + d (matches reshape(B,NQ,H,D)).
// offattn: (MTOT x 96) f32: [0..63] = offsets (h*8 + p*2 + {x,y}),
//                           [64..95] = attn logits (h*4 + p).
// ---------------------------------------------------------------------------
__device__ __forceinline__ float bilerp_corner(const _Float16* vb, float xi,
                                               float yi, float w) {
  bool valid = (xi >= 0.0f) & (xi <= 127.0f) & (yi >= 0.0f) & (yi <= 127.0f);
  int cx = (int)fminf(fmaxf(xi, 0.0f), 127.0f);
  int cy = (int)fminf(fmaxf(yi, 0.0f), 127.0f);
  float g = (float)vb[(size_t)(cy * 128 + cx) * 256];
  return valid ? w * g : 0.0f;
}

__global__ __launch_bounds__(256) void msda_sample(
    const _Float16* __restrict__ value_h, const float* __restrict__ offattn,
    _Float16* __restrict__ out_h) {
  int gid  = blockIdx.x * blockDim.x + threadIdx.x;
  int lane = gid & 31;              // channel d
  int wid  = gid >> 5;              // m*HEADS + h
  int h    = wid & (HEADS - 1);
  int m    = wid >> 3;
  int b    = m >> 14;               // / NQ
  int nq   = m & (NQ - 1);

  float refx = (float)(nq & 127) * (1.0f / 127.0f);
  float refy = (float)(nq >> 7)  * (1.0f / 127.0f);

  const float* oa = offattn + (size_t)m * 96;

  // softmax over 4 attention logits for this head
  float l0 = oa[64 + h * 4 + 0], l1 = oa[64 + h * 4 + 1];
  float l2 = oa[64 + h * 4 + 2], l3 = oa[64 + h * 4 + 3];
  float mx = fmaxf(fmaxf(l0, l1), fmaxf(l2, l3));
  float e0 = __expf(l0 - mx), e1 = __expf(l1 - mx);
  float e2 = __expf(l2 - mx), e3 = __expf(l3 - mx);
  float inv = 1.0f / (e0 + e1 + e2 + e3);
  float aw[4] = {e0 * inv, e1 * inv, e2 * inv, e3 * inv};

  const _Float16* vb = value_h + (size_t)b * NQ * 256 + h * 32 + lane;

  float accd = 0.0f;
#pragma unroll
  for (int p = 0; p < POINTS; ++p) {
    float ox = oa[h * 8 + p * 2 + 0];
    float oy = oa[h * 8 + p * 2 + 1];
    float px = refx * 128.0f + ox - 0.5f;   // (ref + off/W) * W - 0.5
    float py = refy * 128.0f + oy - 0.5f;
    float x0 = floorf(px), y0 = floorf(py);
    float fx = px - x0, fy = py - y0;
    float s = 0.0f;
    s += bilerp_corner(vb, x0,        y0,        (1.0f - fx) * (1.0f - fy));
    s += bilerp_corner(vb, x0 + 1.0f, y0,        fx * (1.0f - fy));
    s += bilerp_corner(vb, x0,        y0 + 1.0f, (1.0f - fx) * fy);
    s += bilerp_corner(vb, x0 + 1.0f, y0 + 1.0f, fx * fy);
    accd += aw[p] * s;
  }
  out_h[(size_t)m * 256 + h * 32 + lane] = (_Float16)accd;
}

// ---------------------------------------------------------------------------
// out = x + layernorm(y) * gamma + beta ; one 256-thread block per row.
// ---------------------------------------------------------------------------
__global__ __launch_bounds__(256) void ln_residual(
    const float* __restrict__ y, const float* __restrict__ x,
    const float* __restrict__ gamma, const float* __restrict__ beta,
    float* __restrict__ out) {
  int m = blockIdx.x;
  int c = threadIdx.x;
  size_t base = (size_t)m * 256;
  float v = y[base + c];
  float s = v, sq = v * v;
#pragma unroll
  for (int off = 16; off > 0; off >>= 1) {
    s  += __shfl_xor(s, off, 32);
    sq += __shfl_xor(sq, off, 32);
  }
  __shared__ float ps[8], pq[8];
  int wave = c >> 5, lane = c & 31;
  if (lane == 0) { ps[wave] = s; pq[wave] = sq; }
  __syncthreads();
  if (c == 0) {
    float ts = 0.0f, tq = 0.0f;
#pragma unroll
    for (int i = 0; i < 8; ++i) { ts += ps[i]; tq += pq[i]; }
    ps[0] = ts; pq[0] = tq;
  }
  __syncthreads();
  float mean = ps[0] * (1.0f / 256.0f);
  float var  = pq[0] * (1.0f / 256.0f) - mean * mean;
  float rs   = rsqrtf(var + 1e-5f);
  out[base + c] = x[base + c] + (v - mean) * rs * gamma[c] + beta[c];
}

// ---------------------------------------------------------------------------
// Host-side orchestration
// ---------------------------------------------------------------------------
extern "C" void kernel_launch(void* const* d_in, const int* in_sizes, int n_in,
                              void* d_out, int out_size, void* d_ws,
                              size_t ws_size, hipStream_t stream) {
  const float* q      = (const float*)d_in[0];
  const float* W_off  = (const float*)d_in[1];
  const float* b_off  = (const float*)d_in[2];
  const float* W_attn = (const float*)d_in[3];
  const float* b_attn = (const float*)d_in[4];
  const float* W_val  = (const float*)d_in[5];
  const float* b_val  = (const float*)d_in[6];
  const float* W_out  = (const float*)d_in[7];
  const float* b_out  = (const float*)d_in[8];
  const float* W1     = (const float*)d_in[9];
  const float* b1     = (const float*)d_in[10];
  const float* W2     = (const float*)d_in[11];
  const float* b2     = (const float*)d_in[12];
  const float* gamma  = (const float*)d_in[13];
  const float* beta   = (const float*)d_in[14];
  float* out = (float*)d_out;

  char* ws = (char*)d_ws;
  const size_t M = MTOT;
  // Workspace layout (bytes)
  size_t o = 0;
  _Float16* q_h   = (_Float16*)(ws + o); o += M * 256 * 2;      // 16 MB
  _Float16* val_h = (_Float16*)(ws + o); o += M * 256 * 2;      // 16 MB
  float*    offat = (float*)   (ws + o); o += M * 96  * 4;      // 12.6 MB
  _Float16* att_h = (_Float16*)(ws + o); o += M * 256 * 2;      // 16 MB
  float*    x_f   = (float*)   (ws + o); o += M * 256 * 4;      // 33.6 MB
  _Float16* x_h   = (_Float16*)(ws + o); o += M * 256 * 2;      // 16 MB
  _Float16* h_ffn = (_Float16*)(ws + o); o += M * 512 * 2;      // 33.6 MB
  _Float16* wcat  = (_Float16*)(ws + o); o += NPROJ * 256 * 2;
  _Float16* woutT = (_Float16*)(ws + o); o += 256 * 256 * 2;
  _Float16* w1T   = (_Float16*)(ws + o); o += 512 * 256 * 2;
  _Float16* w2T   = (_Float16*)(ws + o); o += 256 * 512 * 2;
  float*    b352  = (float*)   (ws + o); o += 352 * 4;
  // y reuses the dead q_h + val_h region (32 MB needed, 32 MB available)
  float* y = (float*)ws;

  // 1) convert q -> f16
  {
    int n = (int)(M * 256);
    cvt_f32_to_f16<<<(n + 255) / 256, 256, 0, stream>>>(q, q_h, n);
  }
  // 2) transpose + convert weights (dst is N-major so B frags load like A frags)
  transpose_cvt<<<(256 * 256 + 255) / 256, 256, 0, stream>>>(W_val,  wcat,             256, 256);
  transpose_cvt<<<(256 * 64  + 255) / 256, 256, 0, stream>>>(W_off,  wcat + 256 * 256, 256, 64);
  transpose_cvt<<<(256 * 32  + 255) / 256, 256, 0, stream>>>(W_attn, wcat + 320 * 256, 256, 32);
  transpose_cvt<<<(256 * 256 + 255) / 256, 256, 0, stream>>>(W_out,  woutT, 256, 256);
  transpose_cvt<<<(256 * 512 + 255) / 256, 256, 0, stream>>>(W1,     w1T,   256, 512);
  transpose_cvt<<<(512 * 256 + 255) / 256, 256, 0, stream>>>(W2,     w2T,   512, 256);
  pack_bias<<<2, 256, 0, stream>>>(b_val, b_off, b_attn, b352);

  dim3 blk(256);
  // 3) fused projection GEMM: [value | off | attn_logits]
  gemm_wmma<256, 0><<<dim3(NPROJ / 32, M / 256), blk, 0, stream>>>(
      q_h, wcat, b352, NPROJ, nullptr, val_h, offat);
  // 4) deformable bilinear sampling
  msda_sample<<<(int)(M * HEADS * 32 / 256), blk, 0, stream>>>(val_h, offat, att_h);
  // 5) output projection + residuals: x = out@W_out + b_out + 2q
  gemm_wmma<256, 1><<<dim3(256 / 32, M / 256), blk, 0, stream>>>(
      att_h, woutT, b_out, 256, q, x_h, x_f);
  // 6) FFN1 + ReLU
  gemm_wmma<256, 2><<<dim3(512 / 32, M / 256), blk, 0, stream>>>(
      x_h, w1T, b1, 512, nullptr, h_ffn, nullptr);
  // 7) FFN2
  gemm_wmma<512, 3><<<dim3(256 / 32, M / 256), blk, 0, stream>>>(
      h_ffn, w2T, b2, 256, nullptr, nullptr, y);
  // 8) layernorm + outer residual
  ln_residual<<<(int)M, blk, 0, stream>>>(y, x_f, gamma, beta, out);

  (void)in_sizes; (void)n_in; (void)out_size; (void)ws_size;
}
